// StackModel_41102837022758
// MI455X (gfx1250) — matrix operand order repaired and checked by reading, not compiled, not run on hardware
//
#include <hip/hip_runtime.h>

#define T_STEPS 256
#define BATCH   64
#define DIN     256
#define HID     512
#define OUTD    256
#define NSTACK  12

typedef __bf16 v16bf __attribute__((ext_vector_type(16)));
typedef float  v8f   __attribute__((ext_vector_type(8)));

union Frag { v16bf v; unsigned u[8]; };

__device__ inline float sigf(float x) { return 1.f / (1.f + __expf(-x)); }

// A fragment: 16x32 bf16, rows mBase..mBase+15 of row-major act (ld = HID).
// lane l: M = l&15 ; K-halves: lanes 0-15 own K {0..7,16..23}, lanes 16-31 own {8..15,24..31}.
__device__ inline v16bf loadA(const __bf16* __restrict__ act, int mBase, int k0, int ld, int lane) {
  Frag f;
  const __bf16* row = act + (mBase + (lane & 15)) * ld;
  int kh = (lane >> 4) * 8;
#pragma unroll
  for (int v = 0; v < 8; ++v) {
    int k = k0 + kh + ((v & 4) << 2) + 2 * (v & 3);
    f.u[v] = *(const unsigned*)(row + k);
  }
  return f.v;
}

// B fragment: 32x16 bf16 where B[k][n] = W[jBase+n][k], W row-major (ldw = K total).
// lane l: N = l&15 ; lanes 0-15 own K {0..7,16..23}, lanes 16-31 own {8..15,24..31} (2 packed per VGPR).
__device__ inline v16bf loadB(const __bf16* __restrict__ W, int jBase, int k0, int ldw, int lane) {
  Frag f;
  const __bf16* row = W + (jBase + (lane & 15)) * ldw;
  int kh = (lane >> 4) * 8;
#pragma unroll
  for (int v = 0; v < 8; ++v) {
    f.u[v] = *(const unsigned*)(row + k0 + kh + ((v & 4) << 2) + 2 * (v & 3));
  }
  return f.v;
}

// ---------------- LSTM cell (both layers): g = inAct@wih^T + hIn@whh^T + b ; fused i/f/g/o ----
// grid 128 x block 128. WG = (mTile 0..3, jTile 0..31); wave w = gate w tile at col w*512+jt*16.
__global__ __launch_bounds__(128) void lstm_step(
    const __bf16* __restrict__ inAct, const __bf16* __restrict__ hIn,
    const __bf16* __restrict__ wih,   const __bf16* __restrict__ whh,
    const float*  __restrict__ bih,   const float*  __restrict__ bhh,
    float* __restrict__ c, __bf16* __restrict__ hOut, __bf16* __restrict__ cOutBf)
{
  __shared__ float tile[4][16][16];
  int lane = threadIdx.x & 31;
  int wave = threadIdx.x >> 5;
  int mBase = (blockIdx.x >> 5) * 16;
  int jt    = blockIdx.x & 31;
  int jBase = wave * HID + jt * 16;

  v8f acc = {};
  for (int kk = 0; kk < HID; kk += 32) {
    v16bf va = loadA(inAct, mBase, kk, HID, lane);
    v16bf vb = loadB(wih,  jBase, kk, HID, lane);
    acc = __builtin_amdgcn_wmma_f32_16x16x32_bf16(false, va, false, vb, (short)0, acc, false, false);
  }
  for (int kk = 0; kk < HID; kk += 32) {
    v16bf va = loadA(hIn, mBase, kk, HID, lane);
    v16bf vb = loadB(whh, jBase, kk, HID, lane);
    acc = __builtin_amdgcn_wmma_f32_16x16x32_bf16(false, va, false, vb, (short)0, acc, false, false);
  }
  int n = lane & 15;
  float bias = bih[jBase + n] + bhh[jBase + n];
#pragma unroll
  for (int v = 0; v < 8; ++v)
    tile[wave][v + ((lane >> 4) << 3)][n] = acc[v] + bias;
  __syncthreads();

  for (int e = threadIdx.x; e < 256; e += 128) {
    int m = e >> 4, nn = e & 15;
    int b = mBase + m, j = jt * 16 + nn;
    float gi = tile[0][m][nn], gf = tile[1][m][nn];
    float gg = tile[2][m][nn], go = tile[3][m][nn];
    float cp = c[b * HID + j];
    float c2 = sigf(gf) * cp + sigf(gi) * tanhf(gg);
    float h  = sigf(go) * tanhf(c2);
    c[b * HID + j] = c2;
    hOut[b * HID + j] = (__bf16)h;
    if (cOutBf) cOutBf[b * HID + j] = (__bf16)c2;
  }
}

// ---------------- Gate hidden layers: hid[g] = relu(c1 @ w1[g]^T + b1[g]) --------------------
__global__ __launch_bounds__(128) void gate_hidden(
    const __bf16* __restrict__ c1bf, const __bf16* __restrict__ w1bf,
    const float* __restrict__ b1_0, const float* __restrict__ b1_1, const float* __restrict__ b1_2,
    __bf16* __restrict__ hid)
{
  int lane = threadIdx.x & 31;
  int tid  = blockIdx.x * 4 + (threadIdx.x >> 5);  // 0..383
  int g    = tid >> 7;
  int rem  = tid & 127;
  int mBase = (rem >> 5) * 16;
  int jBase = (rem & 31) * 16;
  const __bf16* W  = w1bf + g * (HID * HID);
  const float*  b1 = g == 0 ? b1_0 : (g == 1 ? b1_1 : b1_2);

  v8f acc = {};
  for (int kk = 0; kk < HID; kk += 32) {
    v16bf va = loadA(c1bf, mBase, kk, HID, lane);
    v16bf vb = loadB(W,    jBase, kk, HID, lane);
    acc = __builtin_amdgcn_wmma_f32_16x16x32_bf16(false, va, false, vb, (short)0, acc, false, false);
  }
  int n = lane & 15;
  float bias = b1[jBase + n];
#pragma unroll
  for (int v = 0; v < 8; ++v) {
    int row = mBase + v + ((lane >> 4) << 3);
    float xv = acc[v] + bias;
    hid[g * (BATCH * HID) + row * HID + jBase + n] = (__bf16)(xv > 0.f ? xv : 0.f);
  }
}

// ---------------- Stack / pointer update + build next xcat ----------------------------------
__global__ __launch_bounds__(512) void stack_step_kernel(
    const float* __restrict__ x, const __bf16* __restrict__ hid,
    const float* __restrict__ w2_0, const float* __restrict__ w2_1, const float* __restrict__ w2_2,
    const float* __restrict__ b2_0, const float* __restrict__ b2_1, const float* __restrict__ b2_2,
    float* __restrict__ stack, float* __restrict__ ptr, __bf16* __restrict__ xcat, int t)
{
  __shared__ float gatesS[3][BATCH];
  __shared__ float ptrOld[BATCH][NSTACK];
  __shared__ float ptrNew[BATCH][NSTACK];
  int tid = threadIdx.x, lane = tid & 31, wave = tid >> 5;

  // 192 dot products (3 gates x 64 batch), 12 per wave, wave-reduced
  for (int i = 0; i < 12; ++i) {
    int d = wave * 12 + i;
    int g = d >> 6, b = d & 63;
    const float*  w2   = g == 0 ? w2_0 : (g == 1 ? w2_1 : w2_2);
    const __bf16* hrow = hid + g * (BATCH * HID) + b * HID;
    float s = 0.f;
    for (int k = lane; k < HID; k += 32) s += (float)hrow[k] * w2[k];
    for (int off = 16; off; off >>= 1) s += __shfl_xor(s, off);
    if (lane == 0) {
      const float* b2 = g == 0 ? b2_0 : (g == 1 ? b2_1 : b2_2);
      gatesS[g][b] = sigf(s + b2[0]);
    }
  }
  for (int i = tid; i < BATCH * NSTACK; i += 512) ptrOld[i / NSTACK][i % NSTACK] = ptr[i];
  __syncthreads();

  if (tid < BATCH) {
    int b = tid;
    float pu = gatesS[0][b], po = gatesS[1][b], pn = gatesS[2][b];
    float np[NSTACK]; float sum = 0.f;
    for (int n = 0; n < NSTACK; ++n) {
      float up = ptrOld[b][(n + NSTACK - 1) % NSTACK];
      float dn = ptrOld[b][(n + 1) % NSTACK];
      float v  = pu * up + po * dn + pn * ptrOld[b][n];
      v = v > 0.f ? v : 0.f;
      float v2 = v * v;
      v = v2 * v2 * v;               // relu(x)^5
      np[n] = v; sum += v;
    }
    float inv = 1.f / (sum + 1e-8f);
    for (int n = 0; n < NSTACK; ++n) {
      float v = np[n] * inv;
      ptrNew[b][n] = v; ptr[b * NSTACK + n] = v;
    }
  }
  __syncthreads();

  for (int idx = tid; idx < BATCH * NSTACK * DIN; idx += 512) {
    int b = idx / (NSTACK * DIN);
    int r = idx - b * (NSTACK * DIN);
    int n = r / DIN, d = r - n * DIN;
    float st  = stack[idx];
    float pu  = ptrOld[b][(n + NSTACK - 1) % NSTACK];
    float pc  = ptrOld[b][n];
    float val = x[(size_t)b * T_STEPS * DIN + (size_t)t * DIN + d];
    float spush = st * (1.f - pu) + val * pu;
    float spop  = st * (1.f - pc) + 0.001f * pc;
    stack[idx]  = gatesS[0][b] * spush + gatesS[1][b] * spop + gatesS[2][b] * st;
  }
  __threadfence();
  __syncthreads();

  if (t + 1 < T_STEPS) {
    for (int idx = tid; idx < BATCH * DIN; idx += 512) {
      int b = idx / DIN, d = idx - b * DIN;
      float s = 0.f;
      for (int n = 0; n < NSTACK; ++n) s += ptrNew[b][n] * stack[(b * NSTACK + n) * DIN + d];
      xcat[b * HID + DIN + d] = (__bf16)s;
      xcat[b * HID + d] = (__bf16)x[(size_t)b * T_STEPS * DIN + (size_t)(t + 1) * DIN + d];
    }
  }
}

// ---------------- Per-step fc: out[b,t,:] = h2_t @ fc_w^T + fc_b ----------------------------
__global__ __launch_bounds__(128) void fc_step(
    const __bf16* __restrict__ h2bf, const __bf16* __restrict__ fcwbf,
    const float* __restrict__ fcb, float* __restrict__ out, int t)
{
  int lane = threadIdx.x & 31;
  int tile = blockIdx.x * 4 + (threadIdx.x >> 5);  // 0..63
  int mBase = (tile >> 4) * 16;
  int jBase = (tile & 15) * 16;
  v8f acc = {};
  for (int kk = 0; kk < HID; kk += 32) {
    v16bf va = loadA(h2bf,  mBase, kk, HID, lane);
    v16bf vb = loadB(fcwbf, jBase, kk, HID, lane);
    acc = __builtin_amdgcn_wmma_f32_16x16x32_bf16(false, va, false, vb, (short)0, acc, false, false);
  }
  int n = lane & 15;
  float bias = fcb[jBase + n];
#pragma unroll
  for (int v = 0; v < 8; ++v) {
    int b = mBase + v + ((lane >> 4) << 3);
    out[(size_t)b * T_STEPS * OUTD + (size_t)t * OUTD + jBase + n] = acc[v] + bias;
  }
}

// ---------------- Init helpers ---------------------------------------------------------------
__global__ void conv_bf(const float* __restrict__ src, __bf16* __restrict__ dst, int n) {
  int i = blockIdx.x * blockDim.x + threadIdx.x;
  if (i < n) dst[i] = (__bf16)src[i];
}

__global__ void init_state(float* c1, float* c2, __bf16* h1a, __bf16* h1b,
                           __bf16* h2a, __bf16* h2b, float* stack, float* ptr,
                           __bf16* xcat, const float* __restrict__ x) {
  int i = blockIdx.x * blockDim.x + threadIdx.x;
  if (i < BATCH * HID) {
    c1[i] = 0.f; c2[i] = 0.f;
    h1a[i] = (__bf16)0.f; h1b[i] = (__bf16)0.f;
    h2a[i] = (__bf16)0.f; h2b[i] = (__bf16)0.f;
  }
  if (i < BATCH * NSTACK * DIN) stack[i] = 0.001f;   // stack0 = broadcast zero_vec (a=b=0.001)
  if (i < BATCH * NSTACK) ptr[i] = (i % NSTACK == 0) ? 1.f : 0.f;
  if (i < BATCH * DIN) {
    int b = i / DIN, d = i - b * DIN;
    xcat[b * HID + d]       = (__bf16)x[(size_t)b * T_STEPS * DIN + d]; // x_0
    xcat[b * HID + DIN + d] = (__bf16)0.001f;                           // s_0 = 0.001 exactly
  }
}

// ---------------- Host side ------------------------------------------------------------------
extern "C" void kernel_launch(void* const* d_in, const int* in_sizes, int n_in,
                              void* d_out, int out_size, void* d_ws, size_t ws_size,
                              hipStream_t stream) {
  (void)in_sizes; (void)n_in; (void)out_size; (void)ws_size;
  const float* x     = (const float*)d_in[0];
  const float* w_ih1 = (const float*)d_in[1];
  const float* w_hh1 = (const float*)d_in[2];
  const float* bih1  = (const float*)d_in[3];
  const float* bhh1  = (const float*)d_in[4];
  const float* w_ih2 = (const float*)d_in[5];
  const float* w_hh2 = (const float*)d_in[6];
  const float* bih2  = (const float*)d_in[7];
  const float* bhh2  = (const float*)d_in[8];
  const float* gw1f[3] = { (const float*)d_in[9],  (const float*)d_in[13], (const float*)d_in[17] };
  const float* gb1[3]  = { (const float*)d_in[10], (const float*)d_in[14], (const float*)d_in[18] };
  const float* gw2[3]  = { (const float*)d_in[11], (const float*)d_in[15], (const float*)d_in[19] };
  const float* gb2[3]  = { (const float*)d_in[12], (const float*)d_in[16], (const float*)d_in[20] };
  const float* fcwf = (const float*)d_in[21];
  const float* fcb  = (const float*)d_in[22];

  char* ws = (char*)d_ws;
  size_t off = 0;
  auto alloc = [&](size_t bytes) {
    off = (off + 255) & ~(size_t)255;
    void* p = ws + off; off += bytes; return p;
  };
  const int WN = 4 * HID * HID;  // 2048*512 weight elems
  __bf16* wih1 = (__bf16*)alloc(WN * 2);
  __bf16* whh1 = (__bf16*)alloc(WN * 2);
  __bf16* wih2 = (__bf16*)alloc(WN * 2);
  __bf16* whh2 = (__bf16*)alloc(WN * 2);
  __bf16* gw1  = (__bf16*)alloc((size_t)3 * HID * HID * 2);
  __bf16* fcw  = (__bf16*)alloc((size_t)OUTD * HID * 2);
  __bf16* h1b[2] = { (__bf16*)alloc(BATCH * HID * 2), (__bf16*)alloc(BATCH * HID * 2) };
  __bf16* h2b[2] = { (__bf16*)alloc(BATCH * HID * 2), (__bf16*)alloc(BATCH * HID * 2) };
  float*  c1   = (float*)alloc(BATCH * HID * 4);
  float*  c2   = (float*)alloc(BATCH * HID * 4);
  __bf16* c1bf = (__bf16*)alloc(BATCH * HID * 2);
  __bf16* hid  = (__bf16*)alloc((size_t)3 * BATCH * HID * 2);
  float*  stk  = (float*)alloc((size_t)BATCH * NSTACK * DIN * 4);
  float*  ptrB = (float*)alloc(BATCH * NSTACK * 4);
  __bf16* xcat = (__bf16*)alloc(BATCH * HID * 2);

  auto cvt = [&](const float* s, __bf16* d, int n) {
    conv_bf<<<(n + 255) / 256, 256, 0, stream>>>(s, d, n);
  };
  cvt(w_ih1, wih1, WN); cvt(w_hh1, whh1, WN);
  cvt(w_ih2, wih2, WN); cvt(w_hh2, whh2, WN);
  cvt(gw1f[0], gw1, HID * HID);
  cvt(gw1f[1], gw1 + HID * HID, HID * HID);
  cvt(gw1f[2], gw1 + 2 * HID * HID, HID * HID);
  cvt(fcwf, fcw, OUTD * HID);

  init_state<<<(BATCH * NSTACK * DIN + 255) / 256, 256, 0, stream>>>(
      c1, c2, h1b[0], h1b[1], h2b[0], h2b[1], stk, ptrB, xcat, x);

  float* out = (float*)d_out;
  for (int t = 0; t < T_STEPS; ++t) {
    int p = t & 1;
    lstm_step<<<128, 128, 0, stream>>>(xcat, h1b[p], wih1, whh1, bih1, bhh1,
                                       c1, h1b[1 - p], c1bf);
    gate_hidden<<<96, 128, 0, stream>>>(c1bf, gw1, gb1[0], gb1[1], gb1[2], hid);
    stack_step_kernel<<<1, 512, 0, stream>>>(x, hid, gw2[0], gw2[1], gw2[2],
                                             gb2[0], gb2[1], gb2[2], stk, ptrB, xcat, t);
    lstm_step<<<128, 128, 0, stream>>>(h1b[1 - p], h2b[p], wih2, whh2, bih2, bhh2,
                                       c2, h2b[1 - p], nullptr);
    fc_step<<<16, 128, 0, stream>>>(h2b[1 - p], fcw, fcb, out, t);
  }
}
